// ManualLSTM_57836029608481
// MI455X (gfx1250) — matrix operand order repaired and checked by reading, not compiled, not run on hardware
//
#include <hip/hip_runtime.h>
#include <hip/hip_bf16.h>

// ---------------------------------------------------------------------------
// LSTM for MI455X (gfx1250): bf16 WMMA (v_wmma_f32_16x16x32_bf16) for both the
// big input-projection GEMM and the 512 sequential recurrent GEMMs.
// Software-pipelined K loop: next K-chunk's fragments are loaded before the
// current wmma cluster so matrix ops overlap memory latency.
// ---------------------------------------------------------------------------

#define SEQ_LEN   512
#define BATCH     64
#define INSZ      1024
#define HID       1024
#define GATES     (4 * HID)

typedef __attribute__((ext_vector_type(16))) __bf16 v16bf;
typedef __attribute__((ext_vector_type(8)))  __bf16 v8bf;
typedef __attribute__((ext_vector_type(8)))  float  v8f;

// Workspace layout (bytes). All offsets 256-aligned.
static const size_t OFF_XBF   = 0;                        // 512*64*1024 * 2  = 64 MiB
static const size_t OFF_WIH   = 67108864;                 // 4096*1024 * 2    = 8 MiB
static const size_t OFF_WHH   = 75497472;                 // 4096*1024 * 2    = 8 MiB
static const size_t OFF_HBF0  = 83886080;                 // 64*1024 * 2      = 128 KiB
static const size_t OFF_HBF1  = 84017152;                 // 64*1024 * 2      = 128 KiB
static const size_t OFF_CBUF  = 84148224;                 // 64*1024 * 4      = 256 KiB
static const size_t OFF_XPROJ = 84410368;                 // 32768*4096 * 4   = 512 MiB

// ----------------------------- helpers -------------------------------------

__device__ __forceinline__ unsigned short f32_to_bf16_bits(float f) {
    unsigned u = __builtin_bit_cast(unsigned, f);
    unsigned r = u + 0x7FFFu + ((u >> 16) & 1u);   // round-to-nearest-even
    return (unsigned short)(r >> 16);
}

__device__ __forceinline__ float sigmoidf_(float x) {
    return 1.0f / (1.0f + __expf(-x));
}

// A-fragment (16x32 bf16, M x K), per ISA 7.12.2:
//   lanes 0-15 / 16-31 both hold M = lane&15
//   halves 0..7  -> K = k0 + (lane>>4)*8 + j
//   halves 8..15 -> K = k0 + 16 + (lane>>4)*8 + j
__device__ __forceinline__ v16bf load_a_frag(const unsigned short* __restrict__ A,
                                             int lda, int m0, int k0, int lane) {
    const int m    = m0 + (lane & 15);
    const int half = lane >> 4;
    const __bf16* p = reinterpret_cast<const __bf16*>(A) + (size_t)m * lda + k0;
    v8bf lo = *reinterpret_cast<const v8bf*>(p + half * 8);
    v8bf hi = *reinterpret_cast<const v8bf*>(p + 16 + half * 8);
    return __builtin_shufflevector(lo, hi, 0,1,2,3,4,5,6,7,8,9,10,11,12,13,14,15);
}

// B-fragment (32x16 bf16, K x N) built from W stored row-major [N, K]
// (B[k][n] = W[n][k]): lane holds column n = lane&15, K = (lane>>4)*8 + 0..7, 16 + (lane>>4)*8 + 0..7.
__device__ __forceinline__ v16bf load_b_frag(const unsigned short* __restrict__ W,
                                             int ldw, int n0, int k0, int lane) {
    const int n    = n0 + (lane & 15);
    const int half = lane >> 4;
    const __bf16* p = reinterpret_cast<const __bf16*>(W) + (size_t)n * ldw + k0;
    v8bf lo = *reinterpret_cast<const v8bf*>(p + half * 8);
    v8bf hi = *reinterpret_cast<const v8bf*>(p + 16 + half * 8);
    return __builtin_shufflevector(lo, hi, 0,1,2,3,4,5,6,7,8,9,10,11,12,13,14,15);
}

#define WMMA_BF16(a, b, c) \
    __builtin_amdgcn_wmma_f32_16x16x32_bf16(false, (a), false, (b), (short)0, (c), false, false)

// ----------------------------- kernels -------------------------------------

__global__ void cvt_f32_bf16(const float* __restrict__ in,
                             unsigned short* __restrict__ out, size_t n) {
    size_t i = (size_t)blockIdx.x * blockDim.x + threadIdx.x;
    if (i < n) out[i] = f32_to_bf16_bits(in[i]);
}

__global__ void init_state(const float* __restrict__ h0, const float* __restrict__ c0,
                           unsigned short* __restrict__ h_bf, float* __restrict__ c_buf,
                           int n) {
    int i = blockIdx.x * blockDim.x + threadIdx.x;
    if (i < n) {
        h_bf[i]  = f32_to_bf16_bits(h0[i]);
        c_buf[i] = c0[i];
    }
}

// x_proj[m, n] = sum_k X[m,k] * Wih[n,k] + (bias_ih[n] + bias_hh[n])
// M = SEQ*BATCH = 32768, N = 4096, K = 1024. One wave: 1 M-tile x 4 N-tiles.
__global__ __launch_bounds__(128)
void xproj_gemm(const unsigned short* __restrict__ X,
                const unsigned short* __restrict__ Wih,
                const float* __restrict__ b_ih, const float* __restrict__ b_hh,
                float* __restrict__ xproj) {
    const int N = GATES, K = INSZ;
    const int lane = threadIdx.x & 31;
    const int wid  = blockIdx.x * 4 + (threadIdx.x >> 5);
    const int ngroups = N / 64;                 // 64
    const int mtile = wid / ngroups;            // 0..2047
    const int ng    = wid % ngroups;            // 0..63
    const int m0 = mtile * 16;
    const int n0 = ng * 64;

    v8f acc0, acc1, acc2, acc3;
#pragma unroll
    for (int v = 0; v < 8; ++v) { acc0[v] = 0.f; acc1[v] = 0.f; acc2[v] = 0.f; acc3[v] = 0.f; }

    // Prologue: fragments for k0 = 0.
    v16bf a  = load_a_frag(X, K, m0, 0, lane);
    v16bf b0 = load_b_frag(Wih, K, n0 +  0, 0, lane);
    v16bf b1 = load_b_frag(Wih, K, n0 + 16, 0, lane);
    v16bf b2 = load_b_frag(Wih, K, n0 + 32, 0, lane);
    v16bf b3 = load_b_frag(Wih, K, n0 + 48, 0, lane);

    for (int k0 = 0; k0 < K; k0 += 32) {
        const int kn = (k0 + 32 < K) ? (k0 + 32) : 0;   // wrap on last iter (dead result)
        // Issue next chunk's loads before this chunk's wmmas.
        v16bf an  = load_a_frag(X, K, m0, kn, lane);
        v16bf bn0 = load_b_frag(Wih, K, n0 +  0, kn, lane);
        v16bf bn1 = load_b_frag(Wih, K, n0 + 16, kn, lane);
        v16bf bn2 = load_b_frag(Wih, K, n0 + 32, kn, lane);
        v16bf bn3 = load_b_frag(Wih, K, n0 + 48, kn, lane);

        acc0 = WMMA_BF16(a, b0, acc0);
        acc1 = WMMA_BF16(a, b1, acc1);
        acc2 = WMMA_BF16(a, b2, acc2);
        acc3 = WMMA_BF16(a, b3, acc3);

        a = an; b0 = bn0; b1 = bn1; b2 = bn2; b3 = bn3;
    }

    const int half = lane >> 4;
    const int nn   = lane & 15;
    const v8f* accs[4] = { &acc0, &acc1, &acc2, &acc3 };
#pragma unroll
    for (int j = 0; j < 4; ++j) {
        const int n = n0 + j * 16 + nn;
        const float bias = b_ih[n] + b_hh[n];
#pragma unroll
        for (int v = 0; v < 8; ++v) {
            const int m = m0 + v + 8 * half;
            xproj[(size_t)m * N + n] = (*accs[j])[v] + bias;
        }
    }
}

// One recurrent step, fully fused: gates = xg + h_prev @ Whh^T, activations,
// c/h update. One wave owns the same 16x16 (batch x hidden) tile across all
// 4 gates. Grid: 64 blocks x 128 threads = 256 waves (4 mtiles x 64 htiles).
__global__ __launch_bounds__(128)
void lstm_step(const float* __restrict__ xg,            // [BATCH, 4H] for this t
               const unsigned short* __restrict__ Whh,  // [4H, H] bf16
               const unsigned short* __restrict__ h_in, // [BATCH, H] bf16
               unsigned short* __restrict__ h_out,      // [BATCH, H] bf16
               float* __restrict__ c_buf,               // [BATCH, H] f32
               float* __restrict__ out_t) {             // [BATCH, H] f32
    const int lane = threadIdx.x & 31;
    const int wid  = blockIdx.x * 4 + (threadIdx.x >> 5);
    const int htiles = HID / 16;                 // 64
    const int mtile = wid / htiles;              // 0..3
    const int ht    = wid % htiles;              // 0..63
    const int m0 = mtile * 16;
    const int h0 = ht * 16;
    const int half = lane >> 4;
    const int nn   = lane & 15;

    // Seed accumulators with the precomputed input projection.
    v8f acc0, acc1, acc2, acc3;
#pragma unroll
    for (int v = 0; v < 8; ++v) {
        const int m = m0 + v + 8 * half;
        const size_t row = (size_t)m * GATES + h0 + nn;
        acc0[v] = xg[row + 0 * HID];
        acc1[v] = xg[row + 1 * HID];
        acc2[v] = xg[row + 2 * HID];
        acc3[v] = xg[row + 3 * HID];
    }

    // Prologue fragments (k0 = 0).
    v16bf a  = load_a_frag(h_in, HID, m0, 0, lane);
    v16bf b0 = load_b_frag(Whh, HID, 0 * HID + h0, 0, lane);
    v16bf b1 = load_b_frag(Whh, HID, 1 * HID + h0, 0, lane);
    v16bf b2 = load_b_frag(Whh, HID, 2 * HID + h0, 0, lane);
    v16bf b3 = load_b_frag(Whh, HID, 3 * HID + h0, 0, lane);

    for (int k0 = 0; k0 < HID; k0 += 32) {
        const int kn = (k0 + 32 < HID) ? (k0 + 32) : 0;
        v16bf an  = load_a_frag(h_in, HID, m0, kn, lane);
        v16bf bn0 = load_b_frag(Whh, HID, 0 * HID + h0, kn, lane);
        v16bf bn1 = load_b_frag(Whh, HID, 1 * HID + h0, kn, lane);
        v16bf bn2 = load_b_frag(Whh, HID, 2 * HID + h0, kn, lane);
        v16bf bn3 = load_b_frag(Whh, HID, 3 * HID + h0, kn, lane);

        acc0 = WMMA_BF16(a, b0, acc0);
        acc1 = WMMA_BF16(a, b1, acc1);
        acc2 = WMMA_BF16(a, b2, acc2);
        acc3 = WMMA_BF16(a, b3, acc3);

        a = an; b0 = bn0; b1 = bn1; b2 = bn2; b3 = bn3;
    }

    // Elementwise LSTM cell update on the accumulator tile.
#pragma unroll
    for (int v = 0; v < 8; ++v) {
        const int b = m0 + v + 8 * half;   // batch index
        const int h = h0 + nn;             // hidden index
        const size_t idx = (size_t)b * HID + h;
        const float ig = sigmoidf_(acc0[v]);
        const float fg = sigmoidf_(acc1[v]);
        const float gg = tanhf(acc2[v]);
        const float og = sigmoidf_(acc3[v]);
        const float c  = fg * c_buf[idx] + ig * gg;
        const float hn = og * tanhf(c);
        c_buf[idx]  = c;
        out_t[idx]  = hn;
        h_out[idx]  = f32_to_bf16_bits(hn);
    }
}

__global__ void tail_copy(const float* __restrict__ last_h,
                          const float* __restrict__ c_buf,
                          float* __restrict__ h_last, float* __restrict__ c_last,
                          int n) {
    int i = blockIdx.x * blockDim.x + threadIdx.x;
    if (i < n) {
        h_last[i] = last_h[i];
        c_last[i] = c_buf[i];
    }
}

// ----------------------------- launcher ------------------------------------

extern "C" void kernel_launch(void* const* d_in, const int* in_sizes, int n_in,
                              void* d_out, int out_size, void* d_ws, size_t ws_size,
                              hipStream_t stream) {
    const float* x    = (const float*)d_in[0];   // [T, B, I]
    const float* wih  = (const float*)d_in[1];   // [4H, I]
    const float* whh  = (const float*)d_in[2];   // [4H, H]
    const float* b_ih = (const float*)d_in[3];   // [4H]
    const float* b_hh = (const float*)d_in[4];   // [4H]
    const float* h0   = (const float*)d_in[5];   // [B, H]
    const float* c0   = (const float*)d_in[6];   // [B, H]
    float* out = (float*)d_out;                  // outputs[T*B*H] ++ h_last ++ c_last

    char* ws = (char*)d_ws;
    unsigned short* x_bf   = (unsigned short*)(ws + OFF_XBF);
    unsigned short* wih_bf = (unsigned short*)(ws + OFF_WIH);
    unsigned short* whh_bf = (unsigned short*)(ws + OFF_WHH);
    unsigned short* hbf0   = (unsigned short*)(ws + OFF_HBF0);
    unsigned short* hbf1   = (unsigned short*)(ws + OFF_HBF1);
    float*          c_buf  = (float*)(ws + OFF_CBUF);
    float*          xproj  = (float*)(ws + OFF_XPROJ);

    const size_t nx = (size_t)SEQ_LEN * BATCH * INSZ;      // 33554432
    const size_t nw = (size_t)GATES * INSZ;                // 4194304
    const int    ns = BATCH * HID;                         // 65536

    cvt_f32_bf16<<<(unsigned)((nx + 255) / 256), 256, 0, stream>>>(x,   x_bf,   nx);
    cvt_f32_bf16<<<(unsigned)((nw + 255) / 256), 256, 0, stream>>>(wih, wih_bf, nw);
    cvt_f32_bf16<<<(unsigned)((nw + 255) / 256), 256, 0, stream>>>(whh, whh_bf, nw);
    init_state<<<(ns + 255) / 256, 256, 0, stream>>>(h0, c0, hbf0, c_buf, ns);

    // x_proj: M=32768 (2048 mtiles) x N=4096 (64 n-groups of 64) -> 131072 waves
    xproj_gemm<<<32768, 128, 0, stream>>>(x_bf, wih_bf, b_ih, b_hh, xproj);

    for (int t = 0; t < SEQ_LEN; ++t) {
        const unsigned short* hin = (t & 1) ? hbf1 : hbf0;
        unsigned short*       hout = (t & 1) ? hbf0 : hbf1;
        lstm_step<<<64, 128, 0, stream>>>(
            xproj + (size_t)t * BATCH * GATES, whh_bf, hin, hout, c_buf,
            out + (size_t)t * BATCH * HID);
    }

    tail_copy<<<(ns + 255) / 256, 256, 0, stream>>>(
        out + (size_t)(SEQ_LEN - 1) * BATCH * HID, c_buf,
        out + (size_t)SEQ_LEN * BATCH * HID,
        out + (size_t)SEQ_LEN * BATCH * HID + ns, ns);
}
